// DAttn_48447231099419
// MI455X (gfx1250) — matrix operand order repaired and checked by reading, not compiled, not run on hardware
//
#include <hip/hip_runtime.h>

// ---------------------------------------------------------------------------
// Deformable attention for gfx1250 (MI455X).
// All GEMM-shaped stages run on v_wmma_f32_16x16x32_bf16 (f32 accumulate).
// All GEMM operands are laid out K-contiguous so fragments are b128 loads.
// The shared 16x256 A-slice per block is staged into LDS with the Tensor
// Data Mover (tensor_load_to_lds + s_wait_tensorcnt), read back as ds b128.
// ---------------------------------------------------------------------------

typedef __attribute__((ext_vector_type(16))) __bf16       v16bf;
typedef __attribute__((ext_vector_type(8)))  float        v8f;
typedef __attribute__((ext_vector_type(4)))  unsigned int v4u;
typedef __attribute__((ext_vector_type(8)))  int          v8i;
typedef __attribute__((ext_vector_type(4)))  int          v4i;

static constexpr int kB    = 16;     // batch
static constexpr int kC    = 256;    // channels (= K for all dense GEMMs)
static constexpr int kHW   = 4096;   // 64*64
static constexpr int kHR   = 15;     // sampled grid side
static constexpr int kNP   = 225;    // sampled points
static constexpr int kNPAD = 256;    // padded point dim
static constexpr int kHPAD = 240;    // conv output cols padded to 15*16
static constexpr int kKC   = 6400;   // im2col K = 256*25

// ---------------------------------------------------------------------------
// Generic K=256 WMMA GEMM.  Block = 8 waves; block computes one 16-row A
// slice x 8 adjacent 16-col B tiles.  A slice (16x256 bf16 = 8KB) staged in
// LDS via TDM.  A: row-major, row stride 256, k contiguous.  B: element
// (k,n) at B[n*256 + k] (k contiguous).  D: element (m,n) at D[m*sDm+n*sDn].
// ---------------------------------------------------------------------------
template <int OUTMODE, bool HASBIAS>   // OUTMODE: 0 = f32 out, 1 = bf16 out
__global__ void __launch_bounds__(256)
gemm256(const __bf16* __restrict__ A, const __bf16* __restrict__ B,
        const float* __restrict__ bias, void* __restrict__ D,
        int Mt, int NtG,
        long long bA, long long bB, long long bD, int sDm, int sDn) {
  __shared__ __align__(16) __bf16 ldsA[16 * 256];

  int ntg = blockIdx.x % NtG;
  int mt  = (blockIdx.x / NtG) % Mt;
  int b   = blockIdx.x / (NtG * Mt);
  const __bf16* Ab = A + bA * b + (long long)mt * 16 * 256;

#if __has_builtin(__builtin_amdgcn_tensor_load_to_lds) && \
    __has_builtin(__builtin_amdgcn_s_wait_tensorcnt)
  if ((threadIdx.x >> 5) == 0) {          // wave 0 drives the TDM
    unsigned lds_off = (unsigned)(unsigned long long)(const void*)&ldsA[0];
    unsigned long long ga = (unsigned long long)Ab;
    // D# group 0: count=1, lds_addr, global_addr[56:0], type=2
    v4u g0;
    g0.x = 1u;
    g0.y = lds_off;
    g0.z = (unsigned)ga;
    g0.w = (unsigned)((ga >> 32) & 0x01ffffffu) | (2u << 30);
    // D# group 1: 1-D tile of 4096 x 2-byte elements
    v8i g1;
    g1[0] = (int)(1u << 16);                      // data_size = 2B, mask=0
    g1[1] = (int)((4096u & 0xffffu) << 16);       // tensor_dim0 lo16
    g1[2] = (int)((4096u >> 16) | (1u << 16));    // td0 hi16 | tensor_dim1=1
    g1[3] = (int)(4096u << 16);                   // td1 hi16 | tile_dim0=4096
    g1[4] = 0;                                    // tile_dim1=0, tile_dim2=0
    g1[5] = 4096;                                 // tensor_dim0_stride lo32
    g1[6] = 0;
    g1[7] = 0;
    v4i z4 = {0, 0, 0, 0};
    v8i z8 = {0, 0, 0, 0, 0, 0, 0, 0};
    __builtin_amdgcn_tensor_load_to_lds(g0, g1, z4, z4, z8, 0);
    __builtin_amdgcn_s_wait_tensorcnt(0);
  }
  __syncthreads();
#else
  for (int i = threadIdx.x; i < (16 * 256) / 8; i += 256)
    ((v4u*)ldsA)[i] = ((const v4u*)Ab)[i];
  __syncthreads();
#endif

  int wid  = threadIdx.x >> 5;
  int lane = threadIdx.x & 31;
  int n0   = ((ntg << 3) + wid) << 4;

  const __bf16* Bb   = B + bB * b;
  const __bf16* brow = Bb + (long long)(n0 + (lane & 15)) * 256 +
                       ((lane >> 4) << 3);        // this lane's K-row half
  __builtin_prefetch(brow, 0, 1);                 // global_prefetch_b8

  v8f acc = {};
#pragma unroll
  for (int kk = 0; kk < 8; ++kk) {
    int ai = ((lane & 15) << 8) + kk * 32 + ((lane >> 4) << 3);
    union { v16bf v; v4u u[2]; } af, bf;
    af.u[0] = *(const v4u*)&ldsA[ai];             // ds_load_b128
    af.u[1] = *(const v4u*)&ldsA[ai + 16];
    bf.u[0] = *(const v4u*)&brow[kk * 32];        // global_load_b128
    bf.u[1] = *(const v4u*)&brow[kk * 32 + 16];
    acc = __builtin_amdgcn_wmma_f32_16x16x32_bf16(false, af.v, false, bf.v,
                                                  (short)0, acc, false, false);
  }

  int n  = n0 + (lane & 15);
  int mb = (mt << 4) + ((lane >> 4) << 3);
#pragma unroll
  for (int r = 0; r < 8; ++r) {
    int m = mb + r;
    float v = acc[r];
    if (HASBIAS) v += bias[m];
    long long idx = bD * b + (long long)m * sDm + (long long)n * sDn;
    if (OUTMODE == 0) ((float*)D)[idx] = v;
    else              ((__bf16*)D)[idx] = (__bf16)v;
  }
}

// ---------------------------------------------------------------------------
// 5x5 stride-4 conv as implicit-im2col WMMA GEMM (K = 6400 = 200*32).
// A = w1 flat (256 x 6400) bf16, b128 fragment loads.  B gathered from the
// K-contiguous qT (b, hw, c) layout with a branch-free incremental walk:
// per element just two compares + selects + one u16 load + one cndmask.
// ---------------------------------------------------------------------------
__global__ void __launch_bounds__(256)
conv_gemm(const __bf16* __restrict__ w1b, const __bf16* __restrict__ qT,
          const float* __restrict__ b1, float* __restrict__ h) {
  int wave = (int)((blockIdx.x * blockDim.x + threadIdx.x) >> 5);
  int b  = wave / 240;                       // 16 mtiles x 15 ntiles
  int t  = wave % 240;
  int mt = t / 15, nt = t % 15;
  int m0 = mt << 4, n0 = nt << 4;

  int lane  = threadIdx.x & 31;
  int nlane = n0 + (lane & 15);
  bool pv   = nlane < kNP;
  int p  = pv ? nlane : 0;                   // clamp: address always in-bounds
  int py = p / kHR, px = p % kHR;
  const __bf16* qwin = qT + (long long)b * kHW * kC +
                       (long long)((4 * py) * 64 + 4 * px) * kC;

  const __bf16* pa0 = w1b + (long long)(m0 + (lane & 15)) * kKC +
                      ((lane >> 4) << 4);
  const int khalf = (lane >> 4) << 4;
  const __bf16 zero = (__bf16)0.0f;

  v8f acc = {};
  for (int kk = 0; kk < kKC / 32; ++kk) {
    union { v16bf v; v4u u[2]; } af;
    af.u[0] = *(const v4u*)(pa0 + kk * 32);        // global_load_b128
    af.u[1] = *(const v4u*)(pa0 + kk * 32 + 8);

    int kb = kk * 32 + khalf;
    int c  = kb / 25;                              // once per K-tile
    int tt = kb - c * 25;
    int ky = tt / 5;
    int kx = tt - ky * 5;
    const __bf16* pe = qwin + (long long)(ky * 64 + kx) * kC + c;

    v16bf bf;
#pragma unroll
    for (int e = 0; e < 16; ++e) {
      __bf16 vv = *pe;                             // unconditional u16 load
      bf[e] = pv ? vv : zero;                      // value select only
      bool rowEnd = (kx == 4);
      bool chEnd  = rowEnd && (ky == 4);
      pe += chEnd ? (1 - (4 * 64 + 4) * kC)        // next c, back to (0,0)
                  : (rowEnd ? (64 - 4) * kC        // next ky, kx=0
                            : kC);                 // next kx
      kx = rowEnd ? 0 : kx + 1;
      ky = chEnd ? 0 : (rowEnd ? ky + 1 : ky);
    }
    acc = __builtin_amdgcn_wmma_f32_16x16x32_bf16(false, af.v, false, bf,
                                                  (short)0, acc, false, false);
  }

  int n  = n0 + (lane & 15);
  int mb = m0 + ((lane >> 4) << 3);
#pragma unroll
  for (int r = 0; r < 8; ++r) {
    int m = mb + r;
    h[((long long)b * kC + m) * kHPAD + n] = acc[r] + b1[m];
  }
}

// ------------------------- small elementwise kernels ------------------------

__global__ void cvt_kernel(const float* __restrict__ src,
                           __bf16* __restrict__ dst, long long n) {
  long long i = (long long)blockIdx.x * blockDim.x + threadIdx.x;
  if (i < n) dst[i] = (__bf16)src[i];
}

// x (b,c,hw) f32 -> xT (b,hw,c) bf16  (coalesced writes)
__global__ void cvtT_kernel(const float* __restrict__ x,
                            __bf16* __restrict__ xT) {
  long long i = (long long)blockIdx.x * blockDim.x + threadIdx.x;
  int c  = (int)(i & 255);
  long long r = i >> 8;
  int hw = (int)(r & 4095);
  int b  = (int)(r >> 12);
  xT[i] = (__bf16)x[((long long)b * kC + c) * kHW + hw];
}

// GELU -> 1x1 conv to 2 ch -> reference grid -> clip -> pixel coords
__global__ void head_kernel(const float* __restrict__ h,
                            const float* __restrict__ w2,
                            const float* __restrict__ b2,
                            float* __restrict__ dp) {
  int idx = blockIdx.x * blockDim.x + threadIdx.x;
  if (idx >= kB * kNP) return;
  int b = idx / kNP, pp = idx % kNP;
  int py = pp / kHR, px = pp % kHR;
  const float* hb = h + (long long)b * kC * kHPAD + pp;
  float a0 = 0.0f, a1 = 0.0f;
  for (int c = 0; c < kC; ++c) {
    float hv = hb[(long long)c * kHPAD];
    float g  = 0.5f * hv * (1.0f + erff(hv * 0.70710678118654752f));
    a0 += w2[c] * g;            // channel 0 -> y offset
    a1 += w2[kC + c] * g;       // channel 1 -> x offset
  }
  float offy = a0 + b2[0], offx = a1 + b2[1];
  float ry = (py + 0.5f) * (2.0f / 14.0f) - 1.0f;
  float rx = (px + 0.5f) * (2.0f / 14.0f) - 1.0f;
  float dpy = fminf(fmaxf(ry + offy, -1.0f), 1.0f);
  float dpx = fminf(fmaxf(rx + offx, -1.0f), 1.0f);
  long long o = ((long long)b * kNPAD + pp) * 2;
  dp[o + 0] = (dpx + 1.0f) * 0.5f * 63.0f;   // gx pixels
  dp[o + 1] = (dpy + 1.0f) * 0.5f * 63.0f;   // gy pixels
}

// bilinear sample -> xtT (b, p, c) bf16, p padded to 256 with zeros
__global__ void __launch_bounds__(256)
sample_kernel(const float* __restrict__ x, const float* __restrict__ dp,
              __bf16* __restrict__ xtT) {
  int b = blockIdx.x >> 8;          // grid = kB * kNPAD
  int p = blockIdx.x & 255;
  int c = threadIdx.x;
  long long oidx = ((long long)b * kNPAD + p) * kC + c;
  if (p >= kNP) { xtT[oidx] = (__bf16)0.0f; return; }
  long long di = ((long long)b * kNPAD + p) * 2;
  float gx = dp[di + 0], gy = dp[di + 1];
  float x0f = floorf(gx), y0f = floorf(gy);
  float wx1 = gx - x0f, wx0 = 1.0f - wx1;
  float wy1 = gy - y0f, wy0 = 1.0f - wy1;
  int x0 = (int)x0f, y0 = (int)y0f, x1 = x0 + 1, y1 = y0 + 1;
  const float* xb = x + ((long long)b * kC + c) * kHW;
  float acc = 0.0f;
#define GATHER(xi, yi, w)                                                 \
  {                                                                       \
    bool v = ((xi) >= 0) && ((xi) < 64) && ((yi) >= 0) && ((yi) < 64);    \
    int xc = min(max((xi), 0), 63), yc = min(max((yi), 0), 63);           \
    acc += (v ? (w) : 0.0f) * xb[yc * 64 + xc];                           \
  }
  GATHER(x0, y0, wx0 * wy0)
  GATHER(x1, y0, wx1 * wy0)
  GATHER(x0, y1, wx0 * wy1)
  GATHER(x1, y1, wx1 * wy1)
#undef GATHER
  xtT[oidx] = (__bf16)acc;
}

// softmax over n (225 valid of 256); one wave per row, 8 cols/lane
__global__ void __launch_bounds__(256)
softmax_kernel(const float* __restrict__ s, __bf16* __restrict__ attn) {
  long long row = (long long)((blockIdx.x * blockDim.x + threadIdx.x) >> 5);
  int lane = threadIdx.x & 31;
  const float* sr = s + row * kNPAD;
  float v[8];
  float mx = -3.0e38f;
#pragma unroll
  for (int i = 0; i < 8; ++i) {
    int n = lane + i * 32;
    v[i] = (n < kNP) ? sr[n] : -3.0e38f;
    mx = fmaxf(mx, v[i]);
  }
  for (int o = 16; o > 0; o >>= 1) mx = fmaxf(mx, __shfl_xor(mx, o, 32));
  float sum = 0.0f;
#pragma unroll
  for (int i = 0; i < 8; ++i) {
    int n = lane + i * 32;
    v[i] = (n < kNP) ? __expf(v[i] - mx) : 0.0f;
    sum += v[i];
  }
  for (int o = 16; o > 0; o >>= 1) sum += __shfl_xor(sum, o, 32);
  float inv = 1.0f / sum;
  __bf16* ar = attn + row * kNPAD;
#pragma unroll
  for (int i = 0; i < 8; ++i) {
    int n = lane + i * 32;
    ar[n] = (__bf16)(v[i] * inv);   // padded cols exact 0
  }
}

// ---------------------------------- driver ----------------------------------

extern "C" void kernel_launch(void* const* d_in, const int* in_sizes, int n_in,
                              void* d_out, int out_size, void* d_ws,
                              size_t ws_size, hipStream_t stream) {
  (void)in_sizes; (void)n_in; (void)out_size; (void)ws_size;
  const float* x  = (const float*)d_in[0];
  const float* w1 = (const float*)d_in[1];
  const float* b1 = (const float*)d_in[2];
  const float* w2 = (const float*)d_in[3];
  const float* b2 = (const float*)d_in[4];
  const float* wq = (const float*)d_in[5];
  const float* bq = (const float*)d_in[6];
  const float* wk = (const float*)d_in[7];
  const float* bk = (const float*)d_in[8];
  const float* wv = (const float*)d_in[9];
  const float* bv = (const float*)d_in[10];
  float* out = (float*)d_out;

  char* ws = (char*)d_ws;
  size_t off = 0;
  auto carve = [&](size_t bytes) -> char* {
    char* p = ws + off;
    off = (off + bytes + 255) & ~(size_t)255;
    return p;
  };
  __bf16* xT  = (__bf16*)carve((size_t)kB * kHW * kC * 2);    // x^T  bf16
  __bf16* qT  = (__bf16*)carve((size_t)kB * kHW * kC * 2);    // q^T  bf16
  __bf16* wqb = (__bf16*)carve((size_t)kC * kC * 2);
  __bf16* wkb = (__bf16*)carve((size_t)kC * kC * 2);
  __bf16* wvb = (__bf16*)carve((size_t)kC * kC * 2);
  __bf16* w1b = (__bf16*)carve((size_t)kC * kKC * 2);
  float*  h   = (float*) carve((size_t)kB * kC * kHPAD * 4);
  float*  dp  = (float*) carve((size_t)kB * kNPAD * 2 * 4);
  __bf16* xtT = (__bf16*)carve((size_t)kB * kNPAD * kC * 2);  // (b,p,c)
  __bf16* kfT = (__bf16*)carve((size_t)kB * kNPAD * kC * 2);  // (b,p,c)
  __bf16* vf  = (__bf16*)carve((size_t)kB * kC * kNPAD * 2);  // (b,c,p)
  float*  s   = (float*) carve((size_t)kB * kHW * kNPAD * 4);
  __bf16* at  = (__bf16*)carve((size_t)kB * kHW * kNPAD * 2); // (b,hw,n)

  const long long bHWC = (long long)kHW * kC;     // 4096*256
  const long long bPC  = (long long)kNPAD * kC;   // 256*256

  // 0) conversions
  cvtT_kernel<<<(unsigned)(((long long)kB * bHWC) / 256), 256, 0, stream>>>(x, xT);
  cvt_kernel<<<(kC * kC + 255) / 256, 256, 0, stream>>>(wq, wqb, kC * kC);
  cvt_kernel<<<(kC * kC + 255) / 256, 256, 0, stream>>>(wk, wkb, kC * kC);
  cvt_kernel<<<(kC * kC + 255) / 256, 256, 0, stream>>>(wv, wvb, kC * kC);
  cvt_kernel<<<(kC * kKC + 255) / 256, 256, 0, stream>>>(w1, w1b,
                                                         (long long)kC * kKC);

  // 1) q^T[hw][o] = sum_c wq[o][c] xT[hw][c] + bq[o]
  gemm256<1, true><<<kB * 16 * 32, 256, 0, stream>>>(
      wqb, xT, bq, qT, /*Mt*/16, /*NtG*/32,
      /*bA*/0, /*bB*/bHWC, /*bD*/bHWC, /*sDm*/1, /*sDn*/256);

  // 2) offset conv (im2col WMMA): h = w1 (*) q + b1
  conv_gemm<<<(kB * 16 * 15) / 8, 256, 0, stream>>>(w1b, qT, b1, h);

  // 3) GELU + 2-ch head + reference grid + clip -> pixel sample coords
  head_kernel<<<(kB * kNP + 255) / 256, 256, 0, stream>>>(h, w2, b2, dp);

  // 4) bilinear sample -> xtT
  sample_kernel<<<kB * kNPAD, 256, 0, stream>>>(x, dp, xtT);

  // 5) kfT[p][o] = wk @ xt + bk ;  vf[o][p] = wv @ xt + bv
  gemm256<1, true><<<kB * 16 * 2, 256, 0, stream>>>(
      wkb, xtT, bk, kfT, 16, 2, 0, bPC, bPC, /*sDm*/1, /*sDn*/256);
  gemm256<1, true><<<kB * 16 * 2, 256, 0, stream>>>(
      wvb, xtT, bv, vf, 16, 2, 0, bPC, bPC, /*sDm*/256, /*sDn*/1);

  // 6) logits s[hw][n] = sum_c qT[hw][c] kfT[n][c]
  gemm256<0, false><<<kB * 256 * 2, 256, 0, stream>>>(
      qT, kfT, nullptr, s, /*Mt*/256, /*NtG*/2,
      /*bA*/bHWC, /*bB*/bPC, /*bD*/(long long)kHW * kNPAD,
      /*sDm*/kNPAD, /*sDn*/1);

  // 7) softmax over n -> attn bf16 (padded cols zero)
  softmax_kernel<<<(kB * kHW) / 8, 256, 0, stream>>>(s, at);

  // 8) out[c][hw] = sum_n vf[c][n] attn[hw][n]
  gemm256<0, false><<<kB * 16 * 32, 256, 0, stream>>>(
      vf, at, nullptr, out, /*Mt*/16, /*NtG*/32,
      /*bA*/bPC, /*bB*/(long long)kHW * kNPAD, /*bD*/(long long)kC * kHW,
      /*sDm*/kHW, /*sDn*/1);
}
